// BPA_78666620993907
// MI455X (gfx1250) — compile-verified, matrix operand order checked by reading, not run on hardware
//
#include <hip/hip_runtime.h>
#include <hip/hip_bf16.h>

// Problem constants (from reference): B=16, N=1024, D=256, K=256
#define BB   16
#define NN   1024
#define DD   256
#define KK   256
#define ROWS (BB * NN)        // 16384 rows of X / assign
#define MAT  (ROWS * KK)      // 4194304 elements per [B,N,K] tensor

typedef __attribute__((ext_vector_type(16))) __bf16 v16bf;
typedef __attribute__((ext_vector_type(8)))  float  v8f;

union FragBF {
    v16bf v;
    uint4 q[2];   // 32 bytes
};

__device__ __forceinline__ unsigned short f2bf(float f) {
    unsigned int u = __float_as_uint(f);
    u += 0x7FFFu + ((u >> 16) & 1u);     // round-to-nearest-even
    return (unsigned short)(u >> 16);
}

__device__ __forceinline__ float wave_sum32(float s) {
#pragma unroll
    for (int m = 16; m >= 1; m >>= 1) s += __shfl_xor(s, m, 32);
    return s;
}

// ---------------------------------------------------------------------------
// L2-normalize rows of X [rows, 256] -> bf16. One wave (32 lanes) per row,
// 8 contiguous floats per lane. wave32 shuffle reduction.
// ---------------------------------------------------------------------------
__global__ __launch_bounds__(256)
void l2norm_bf16_kernel(const float* __restrict__ X,
                        unsigned short* __restrict__ Xn, int rows) {
    int row  = blockIdx.x * 8 + (threadIdx.x >> 5);
    if (row >= rows) return;
    int lane = threadIdx.x & 31;
    const float4* p = (const float4*)(X + (size_t)row * DD + lane * 8);
    float4 a = p[0], b = p[1];
    float s = a.x*a.x + a.y*a.y + a.z*a.z + a.w*a.w
            + b.x*b.x + b.y*b.y + b.z*b.z + b.w*b.w;
    s = wave_sum32(s);
    float sc = 1.0f / (sqrtf(s) + 1e-8f);
    unsigned short o0 = f2bf(a.x*sc), o1 = f2bf(a.y*sc), o2 = f2bf(a.z*sc), o3 = f2bf(a.w*sc);
    unsigned short o4 = f2bf(b.x*sc), o5 = f2bf(b.y*sc), o6 = f2bf(b.z*sc), o7 = f2bf(b.w*sc);
    uint4 pk;
    pk.x = (unsigned)o0 | ((unsigned)o1 << 16);
    pk.y = (unsigned)o2 | ((unsigned)o3 << 16);
    pk.z = (unsigned)o4 | ((unsigned)o5 << 16);
    pk.w = (unsigned)o6 | ((unsigned)o7 << 16);
    *((uint4*)(Xn + (size_t)row * DD + lane * 8)) = pk;
}

// ---------------------------------------------------------------------------
// Normalize proto [256,256] -> bf16 row-major (pn) AND transposed (pnT).
// sim GEMM consumes pn rows; z GEMM consumes pnT rows: both end up row-major.
// ---------------------------------------------------------------------------
__global__ __launch_bounds__(256)
void proto_norm_kernel(const float* __restrict__ P,
                       unsigned short* __restrict__ pn,
                       unsigned short* __restrict__ pnT) {
    int row  = blockIdx.x * 8 + (threadIdx.x >> 5);   // 0..255
    int lane = threadIdx.x & 31;
    const float4* p = (const float4*)(P + (size_t)row * DD + lane * 8);
    float4 a = p[0], b = p[1];
    float s = a.x*a.x + a.y*a.y + a.z*a.z + a.w*a.w
            + b.x*b.x + b.y*b.y + b.z*b.z + b.w*b.w;
    s = wave_sum32(s);
    float sc = 1.0f / (sqrtf(s) + 1e-8f);
    float v[8] = {a.x*sc, a.y*sc, a.z*sc, a.w*sc, b.x*sc, b.y*sc, b.z*sc, b.w*sc};
#pragma unroll
    for (int i = 0; i < 8; ++i) {
        unsigned short h = f2bf(v[i]);
        int d = lane * 8 + i;
        pn [(size_t)row * DD + d]  = h;
        pnT[(size_t)d   * KK + row] = h;
    }
}

// ---------------------------------------------------------------------------
// C[M=16384, 256] = A_bf16[16384, 256] x B_bf16[256, 256] (B rows = operand
// columns). The full 128 KB B panel is staged into LDS with CDNA5
// GLOBAL_LOAD_ASYNC_TO_LDS_B128 (ASYNCcnt-tracked, no VGPR round trip),
// fenced by s_wait_asynccnt + workgroup barrier.
// One block (8 waves) owns one 16-row M stripe; each wave computes TWO 16x16
// N tiles, reusing its A fragment across two v_wmma_f32_16x16x32_bf16 per
// 32-deep K step (16 WMMA issues per K sweep). Operand chunking follows the
// ISA 16-bit A/B layout: lanes 0-15 hold K {0-7,16-23}, lanes 16-31 hold
// K {8-15,24-31}. mode==1: also write Q = exp(C * invtau) (Sinkhorn seed).
// ---------------------------------------------------------------------------
__global__ __launch_bounds__(256)
void gemm_bf16_wmma_kernel(const unsigned short* __restrict__ A,
                           const unsigned short* __restrict__ Bm,
                           float* __restrict__ C,
                           float* __restrict__ Q,
                           float invtau, int mode) {
    extern __shared__ unsigned short sB[];   // 256*256 bf16 = 131072 bytes
    {
        // LDS segment byte-offset of sB for this thread (addrspace(3) view).
        unsigned lbase =
            (unsigned)(size_t)((__attribute__((address_space(3))) unsigned short*)sB)
            + threadIdx.x * 16u;
        const unsigned short* gbase = Bm + threadIdx.x * 8;   // 16 B per lane
#pragma unroll
        for (int i = 0; i < 32; ++i) {
            unsigned laddr = lbase + (unsigned)i * 4096u;     // 256 lanes * 16 B
            const unsigned short* g = gbase + i * 2048;       // 4096 B / sizeof
            asm volatile("global_load_async_to_lds_b128 %0, %1, off"
                         :: "v"(laddr), "v"(g)
                         : "memory");
        }
        asm volatile("s_wait_asynccnt 0x0" ::: "memory");
    }
    __syncthreads();

    int tile_m = blockIdx.x;              // 0..1023  (one M stripe per block)
    int wave   = threadIdx.x >> 5;        // 0..7
    int tn0    = wave * 2;                // two adjacent N tiles per wave
    int tn1    = tn0 + 1;
    int lane   = threadIdx.x & 31;
    int hf     = lane >> 4;               // lane half selects K sub-chunks
    int idx    = lane & 15;

    const unsigned short* arow  = A  + (size_t)(tile_m * 16 + idx) * DD;
    const unsigned short* brow0 = sB + (size_t)(tn0 * 16 + idx) * DD;
    const unsigned short* brow1 = sB + (size_t)(tn1 * 16 + idx) * DD;

    v8f acc0 = {}, acc1 = {};
#pragma unroll
    for (int kb = 0; kb < DD; kb += 32) {
        int o0 = kb + 8 * hf;
        int o1 = kb + 16 + 8 * hf;
        FragBF fa, fb0, fb1;
        fa.q[0]  = *((const uint4*)(arow + o0));
        fa.q[1]  = *((const uint4*)(arow + o1));
        __builtin_prefetch(arow + kb + 64, 0, 0);   // global_prefetch_b8
        fb0.q[0] = *((const uint4*)(brow0 + o0));
        fb0.q[1] = *((const uint4*)(brow0 + o1));
        fb1.q[0] = *((const uint4*)(brow1 + o0));
        fb1.q[1] = *((const uint4*)(brow1 + o1));
        acc0 = __builtin_amdgcn_wmma_f32_16x16x32_bf16(
                   false, fa.v, false, fb0.v, (short)0, acc0, false, false);
        acc1 = __builtin_amdgcn_wmma_f32_16x16x32_bf16(
                   false, fa.v, false, fb1.v, (short)0, acc1, false, false);
    }

    int cn0   = tn0 * 16 + idx;
    int cn1   = tn1 * 16 + idx;
    int rbase = tile_m * 16 + (hf << 3);  // C/D layout: vgpr r -> M = r + 8*half
#pragma unroll
    for (int r = 0; r < 8; ++r) {
        size_t rowoff = (size_t)(rbase + r) * KK;
        float v0 = acc0[r], v1 = acc1[r];
        C[rowoff + cn0] = v0;
        C[rowoff + cn1] = v1;
        if (mode) {
            Q[rowoff + cn0] = __expf(v0 * invtau);
            Q[rowoff + cn1] = __expf(v1 * invtau);
        }
    }
}

// ---------------------------------------------------------------------------
// Sinkhorn row pass: q[row,:] *= target / (sum_k q[row,k] + eps).
// Wave per row. Optionally emit bf16 copy (final pass feeds the z GEMM).
// ---------------------------------------------------------------------------
__global__ __launch_bounds__(256)
void sinkhorn_row_kernel(float* __restrict__ q, float target,
                         unsigned short* __restrict__ abf) {
    int row  = blockIdx.x * 8 + (threadIdx.x >> 5);
    int lane = threadIdx.x & 31;
    float4* p = (float4*)(q + (size_t)row * KK + lane * 8);
    float4 a = p[0], b = p[1];
    float s = a.x + a.y + a.z + a.w + b.x + b.y + b.z + b.w;
    s = wave_sum32(s);
    float sc = target / (s + 1e-8f);
    a.x *= sc; a.y *= sc; a.z *= sc; a.w *= sc;
    b.x *= sc; b.y *= sc; b.z *= sc; b.w *= sc;
    p[0] = a; p[1] = b;
    if (abf) {
        uint4 pk;
        pk.x = (unsigned)f2bf(a.x) | ((unsigned)f2bf(a.y) << 16);
        pk.y = (unsigned)f2bf(a.z) | ((unsigned)f2bf(a.w) << 16);
        pk.z = (unsigned)f2bf(b.x) | ((unsigned)f2bf(b.y) << 16);
        pk.w = (unsigned)f2bf(b.z) | ((unsigned)f2bf(b.w) << 16);
        *((uint4*)(abf + (size_t)row * KK + lane * 8)) = pk;
    }
}

// ---------------------------------------------------------------------------
// Sinkhorn column pass: per batch b, q[b,:,k] *= target / (sum_n q[b,n,k]+eps)
// One block per batch, thread per column; fully coalesced across k.
// ---------------------------------------------------------------------------
__global__ __launch_bounds__(256)
void sinkhorn_col_kernel(float* __restrict__ q, float target) {
    int b = blockIdx.x;
    int k = threadIdx.x;
    float* base = q + (size_t)b * NN * KK + k;
    float s = 0.0f;
    for (int n = 0; n < NN; ++n) s += base[(size_t)n * KK];
    float sc = target / (s + 1e-8f);
    for (int n = 0; n < NN; ++n) base[(size_t)n * KK] *= sc;
}

// Column sums per (b,k) for the consistency scalar.
__global__ __launch_bounds__(256)
void colsum_kernel(const float* __restrict__ a, float* __restrict__ cs) {
    int b = blockIdx.x;
    int k = threadIdx.x;
    const float* base = a + (size_t)b * NN * KK + k;
    float s = 0.0f;
    for (int n = 0; n < NN; ++n) s += base[(size_t)n * KK];
    cs[b * KK + k] = s;
}

// mean(A B^T over [B,N,N]) = sum_{b,k} colsum_A[b,k]*colsum_B[b,k] / (B*N*N);
// both cross-modal means are identical by symmetry -> scalar = 1 - clip(mean).
__global__ __launch_bounds__(256)
void consistency_kernel(const float* __restrict__ cr,
                        const float* __restrict__ cs,
                        float* __restrict__ out) {
    __shared__ float red[256];
    int k = threadIdx.x;
    float s = 0.0f;
    for (int b = 0; b < BB; ++b) s += cr[b * KK + k] * cs[b * KK + k];
    red[k] = s;
    __syncthreads();
    for (int st = 128; st > 0; st >>= 1) {
        if (k < st) red[k] += red[k + st];
        __syncthreads();
    }
    if (k == 0) {
        float m = red[0] * (1.0f / 16777216.0f);   // / (B*N*N)
        m = fminf(fmaxf(m, 0.0f), 1.0f);
        out[0] = 1.0f - m;
    }
}

// ---------------------------------------------------------------------------
extern "C" void kernel_launch(void* const* d_in, const int* in_sizes, int n_in,
                              void* d_out, int out_size, void* d_ws, size_t ws_size,
                              hipStream_t stream) {
    const float* X[2] = {(const float*)d_in[0], (const float*)d_in[1]};
    const float* P[2] = {(const float*)d_in[2], (const float*)d_in[3]};

    float* out = (float*)d_out;
    float* Z[2]    = {out,            out + (size_t)MAT};
    float* Aout[2] = {out + 2ull*MAT, out + 3ull*MAT};   // assign (in-place Sinkhorn)
    float* S[2]    = {out + 4ull*MAT, out + 5ull*MAT};   // sim
    float* consis  = out + 6ull * MAT;

    char* ws = (char*)d_ws;
    unsigned short* XN[2]  = {(unsigned short*)(ws),
                              (unsigned short*)(ws + (8ull << 20))};
    unsigned short* AB[2]  = {(unsigned short*)(ws + (16ull << 20)),
                              (unsigned short*)(ws + (24ull << 20))};
    unsigned short* pbase  = (unsigned short*)(ws + (32ull << 20));
    unsigned short* PN[2]  = {pbase,             pbase + 2 * 65536};
    unsigned short* PNT[2] = {pbase + 65536,     pbase + 3 * 65536};
    float* CS[2];
    CS[0] = (float*)(pbase + 4 * 65536);
    CS[1] = CS[0] + BB * KK;

    dim3 blk(256);
    const float invtau = 20.0f;           // 1 / 0.05
    const float trow   = 1.0f / (float)NN;
    const float tcol   = 1.0f / (float)KK;

    for (int m = 0; m < 2; ++m) {
        // normalize features + prototypes -> bf16 operands
        l2norm_bf16_kernel<<<ROWS / 8, blk, 0, stream>>>(X[m], XN[m], ROWS);
        proto_norm_kernel<<<KK / 8, blk, 0, stream>>>(P[m], PN[m], PNT[m]);
        // sim = Xn @ Pn^T  (fused exp -> Sinkhorn seed in assign region)
        gemm_bf16_wmma_kernel<<<ROWS / 16, blk, 131072, stream>>>(
            XN[m], PN[m], S[m], Aout[m], invtau, 1);
        // Sinkhorn balancing (initial global normalization cancels in pass 1)
        for (int it = 0; it < 5; ++it) {
            sinkhorn_row_kernel<<<ROWS / 8, blk, 0, stream>>>(Aout[m], trow, nullptr);
            sinkhorn_col_kernel<<<BB, blk, 0, stream>>>(Aout[m], tcol);
        }
        sinkhorn_row_kernel<<<ROWS / 8, blk, 0, stream>>>(Aout[m], 1.0f, AB[m]);
        // z = A @ Pn
        gemm_bf16_wmma_kernel<<<ROWS / 16, blk, 131072, stream>>>(
            AB[m], PNT[m], Z[m], nullptr, 0.0f, 0);
        colsum_kernel<<<BB, blk, 0, stream>>>(Aout[m], CS[m]);
    }
    consistency_kernel<<<1, blk, 0, stream>>>(CS[0], CS[1], consis);
}
